// MultiHeadAttention_64510408786328
// MI455X (gfx1250) — matrix operand order repaired and checked by reading, not compiled, not run on hardware
//
#include <hip/hip_runtime.h>
#include <hip/hip_bf16.h>

// Problem constants (reference: B=2, S=2048, EMB=1024, H=16, DK=DV=64)
#define B_   2
#define S_   2048
#define EMB_ 1024
#define H_   16
#define DK_  64
#define DV_  64
#define NQK  (H_*DK_)   // 1024 = H*DK = H*DV

typedef __attribute__((ext_vector_type(16))) __bf16 v16bf;
typedef __attribute__((ext_vector_type(8)))  float  v8f;

static __device__ __forceinline__ __bf16 cvt_bf(float f) { return (__bf16)f; }

// A-matrix (16x32) k order: element j of the per-lane v16bf -> K index.
// half = lane>>4.  Contiguous order K = 16*half + j, matching the B operands.
static __device__ __forceinline__ int a_k(int j, int half) {
  return 16 * half + j;
}

static __device__ __forceinline__ v8f wmma_bf16(v16bf a, v16bf b, v8f c) {
  // D = A(16x32 bf16) * B(32x16 bf16) + C(16x16 f32)
  return __builtin_amdgcn_wmma_f32_16x16x32_bf16(false, a, false, b, (short)0, c,
                                                 false, false);
}

// CDNA5 async copy: 16 bytes per lane, global -> LDS, tracked by ASYNCcnt.
static __device__ __forceinline__ void async_copy16(const void* g, void* lds) {
  unsigned           laddr = (unsigned)(uintptr_t)lds;          // low 32b = LDS offset
  unsigned long long gaddr = (unsigned long long)(uintptr_t)g;
  asm volatile("global_load_async_to_lds_b128 %0, %1, off"
               :: "v"(laddr), "v"(gaddr) : "memory");
}
static __device__ __forceinline__ void wait_async0() {
  asm volatile("s_wait_asynccnt 0" ::: "memory");
}

// ---------------------------------------------------------------------------
// Projection GEMM: out[M=B*S, N=1024] = bf16( X[M,1024] @ W[1024,1024] + bias )
// Block (256 thr = 8 waves) computes a 128x64 tile; BK=32.
// X (128x32 f32) and W (32x64 f32) k-slices are staged in LDS via
// global_load_async_to_lds_b128.  Each wave owns 16 rows x 64 cols: one A-tile
// reused across 4 WMMAs per k-chunk.
// TR=1 stores V transposed as [B, H, DV, S] so attention loads V B-tiles
// with single contiguous 32B loads.
// ---------------------------------------------------------------------------
template <bool TR>
__global__ __launch_bounds__(256)
void proj_gemm_bf16(const float* __restrict__ X, const float* __restrict__ W,
                    const float* __restrict__ bias, __bf16* __restrict__ out) {
  __shared__ float Xs[128][32];   // 16 KB
  __shared__ float Ws[32][64];    //  8 KB

  const int tid  = threadIdx.x;
  const int lane = tid & 31;
  const int wrow = (tid >> 5) * 16;        // wave's 16-row slice of the 128-row tile
  const int bm   = blockIdx.x >> 4;        // 0..31  (4096/128)
  const int bn   = blockIdx.x & 15;        // 0..15  (1024/64)
  const int row0 = bm * 128;
  const int col0 = bn * 64;
  const int n    = lane & 15;
  const int half = lane >> 4;
  const int m    = lane & 15;

  v8f acc[4] = {};

#pragma unroll 1
  for (int kb = 0; kb < EMB_; kb += 32) {
    // ---- stage X 128x32 tile: 1024 x 16B segments, 4 per thread ----
#pragma unroll
    for (int j = 0; j < 4; ++j) {
      const int i = tid + 256 * j;
      const int r = i >> 3, s = i & 7;
      async_copy16(X + (size_t)(row0 + r) * EMB_ + kb + s * 4, &Xs[r][s * 4]);
    }
    // ---- stage W 32x64 tile: 512 x 16B segments, 2 per thread ----
#pragma unroll
    for (int j = 0; j < 2; ++j) {
      const int i = tid + 256 * j;
      const int r = i >> 4, s = i & 15;
      async_copy16(W + (size_t)(kb + r) * NQK + col0 + s * 4, &Ws[r][s * 4]);
    }
    wait_async0();        // my copies done...
    __syncthreads();      // ...and everyone else's (each waited before arriving)

    v16bf a;
#pragma unroll
    for (int j = 0; j < 16; ++j) a[j] = cvt_bf(Xs[wrow + m][a_k(j, half)]);
#pragma unroll
    for (int t = 0; t < 4; ++t) {
      v16bf bm_;
#pragma unroll
      for (int j = 0; j < 16; ++j) bm_[j] = cvt_bf(Ws[16 * half + j][16 * t + n]);
      acc[t] = wmma_bf16(a, bm_, acc[t]);
    }
    __syncthreads();      // protect LDS against next chunk's async writes (WAR)
  }

#pragma unroll
  for (int t = 0; t < 4; ++t) {
    const float bb = bias[col0 + 16 * t + n];
#pragma unroll
    for (int r = 0; r < 8; ++r) {
      const int    mm  = r + 8 * half;
      const int    row = row0 + wrow + mm;
      const int    col = col0 + 16 * t + n;
      const __bf16 val = cvt_bf(acc[t][r] + bb);
      if (TR) {
        const int b_ = row >> 11, s_ = row & (S_ - 1);
        const int h_ = col >> 6,  d_ = col & 63;
        out[(((size_t)b_ * H_ + h_) * DV_ + d_) * S_ + s_] = val;
      } else {
        out[(size_t)row * NQK + col] = val;
      }
    }
  }
}

// ---------------------------------------------------------------------------
// Flash attention: one wave per (b, h, 16 q-rows).  kv loop in steps of 32.
// Q,K: [B*S, H*DK] bf16 (dk fast axis -> contiguous K B-tiles)
// Vt : [B, H, DV, S] bf16 (kv fast axis -> contiguous V B-tiles)
// ---------------------------------------------------------------------------
__global__ __launch_bounds__(128)
void flash_attn(const __bf16* __restrict__ Q, const __bf16* __restrict__ K,
                const __bf16* __restrict__ Vt, const unsigned char* __restrict__ mask,
                float* __restrict__ out) {
  __shared__ unsigned short p_lds[4][16][32];   // per-wave P staging (C->A relayout)

  const int lane  = threadIdx.x & 31;
  const int wslot = threadIdx.x >> 5;
  const int wid   = blockIdx.x * 4 + wslot;
  const int qt    = wid % (S_ / 16);
  const int bh    = wid / (S_ / 16);
  const int h     = bh % H_;
  const int b     = bh / H_;
  const int q0    = qt * 16;
  const int n = lane & 15, half = lane >> 4, m = lane & 15;

  // Q A-tiles (16 x 64 split into two 16x32 chunks), loaded once.
  v16bf qa0, qa1;
  const __bf16* qrow = Q + (size_t)(b * S_ + q0 + m) * NQK + h * DK_;
#pragma unroll
  for (int j = 0; j < 16; ++j) {
    qa0[j] = qrow[a_k(j, half)];
    qa1[j] = qrow[32 + a_k(j, half)];
  }

  v8f o0 = {}, o1 = {}, o2 = {}, o3 = {};
  float mi[8], li[8];
#pragma unroll
  for (int r = 0; r < 8; ++r) { mi[r] = -3.0e38f; li[r] = 0.0f; }

  const __bf16* kbase = K + (size_t)(b * S_) * NQK + h * DK_;
  const __bf16* vbase = Vt + ((size_t)b * H_ + h) * DV_ * S_;
  const unsigned char* mrow = mask + ((size_t)b * S_ + q0) * S_;

#pragma unroll 1
  for (int kv = 0; kv < S_; kv += 32) {
    // --- scores: two 16x16 tiles covering kv..kv+31 -------------------------
    const v16bf kb00 = *(const v16bf*)(kbase + (size_t)(kv + n) * NQK + 16 * half);
    const v16bf kb01 = *(const v16bf*)(kbase + (size_t)(kv + n) * NQK + 32 + 16 * half);
    const v16bf kb10 = *(const v16bf*)(kbase + (size_t)(kv + 16 + n) * NQK + 16 * half);
    const v16bf kb11 = *(const v16bf*)(kbase + (size_t)(kv + 16 + n) * NQK + 32 + 16 * half);

    v8f s0 = {}, s1 = {};
    s0 = wmma_bf16(qa0, kb00, s0);
    s0 = wmma_bf16(qa1, kb01, s0);
    s1 = wmma_bf16(qa0, kb10, s1);
    s1 = wmma_bf16(qa1, kb11, s1);

    // --- scale 1/sqrt(64) + mask -------------------------------------------
#pragma unroll
    for (int r = 0; r < 8; ++r) {
      const int mm = r + 8 * half;
      float a = s0[r] * 0.125f;
      float c = s1[r] * 0.125f;
      a = mrow[(size_t)mm * S_ + kv + n]      ? -1.0e9f : a;
      c = mrow[(size_t)mm * S_ + kv + 16 + n] ? -1.0e9f : c;
      s0[r] = a; s1[r] = c;
    }

    // --- online softmax (row stats replicated across the 16-lane group) ----
    float alpha[8];
#pragma unroll
    for (int r = 0; r < 8; ++r) {
      float mx = fmaxf(s0[r], s1[r]);
#pragma unroll
      for (int d = 8; d >= 1; d >>= 1) mx = fmaxf(mx, __shfl_xor(mx, d, 16));
      const float mnew = fmaxf(mi[r], mx);
      const float al   = __expf(mi[r] - mnew);
      const float p0   = __expf(s0[r] - mnew);
      const float p1   = __expf(s1[r] - mnew);
      float rs = p0 + p1;
#pragma unroll
      for (int d = 8; d >= 1; d >>= 1) rs += __shfl_xor(rs, d, 16);
      li[r] = li[r] * al + rs;
      mi[r] = mnew;
      alpha[r] = al;
      s0[r] = p0; s1[r] = p1;
    }
#pragma unroll
    for (int r = 0; r < 8; ++r) {
      o0[r] *= alpha[r]; o1[r] *= alpha[r]; o2[r] *= alpha[r]; o3[r] *= alpha[r];
    }

    // --- P: C-layout -> A-layout via per-wave LDS tile ----------------------
#pragma unroll
    for (int r = 0; r < 8; ++r) {
      const int mm = r + 8 * half;
      p_lds[wslot][mm][n]      = __builtin_bit_cast(unsigned short, cvt_bf(s0[r]));
      p_lds[wslot][mm][16 + n] = __builtin_bit_cast(unsigned short, cvt_bf(s1[r]));
    }
    __builtin_amdgcn_wave_barrier();
    asm volatile("s_wait_dscnt 0" ::: "memory");   // same-wave DS RAW

    v16bf pa;
#pragma unroll
    for (int j = 0; j < 16; ++j)
      pa[j] = __builtin_bit_cast(__bf16, p_lds[wslot][m][a_k(j, half)]);

    // --- O += P @ V  (V B-tiles contiguous thanks to transposed layout) -----
    const v16bf vb0 = *(const v16bf*)(vbase + (size_t)(0 * 16 + n) * S_ + kv + 16 * half);
    const v16bf vb1 = *(const v16bf*)(vbase + (size_t)(1 * 16 + n) * S_ + kv + 16 * half);
    const v16bf vb2 = *(const v16bf*)(vbase + (size_t)(2 * 16 + n) * S_ + kv + 16 * half);
    const v16bf vb3 = *(const v16bf*)(vbase + (size_t)(3 * 16 + n) * S_ + kv + 16 * half);
    o0 = wmma_bf16(pa, vb0, o0);
    o1 = wmma_bf16(pa, vb1, o1);
    o2 = wmma_bf16(pa, vb2, o2);
    o3 = wmma_bf16(pa, vb3, o3);
    __builtin_amdgcn_wave_barrier();   // keep next-iter LDS stores behind reads
  }

  // --- normalize and store fp32 output [B,S,H*DV] ---------------------------
  float* orow = out + (size_t)(b * S_ + q0) * NQK + h * DV_;
#pragma unroll
  for (int r = 0; r < 8; ++r) {
    const int mm  = r + 8 * half;
    const float inv = 1.0f / li[r];
    orow[(size_t)mm * NQK + 0  + n] = o0[r] * inv;
    orow[(size_t)mm * NQK + 16 + n] = o1[r] * inv;
    orow[(size_t)mm * NQK + 32 + n] = o2[r] * inv;
    orow[(size_t)mm * NQK + 48 + n] = o3[r] * inv;
  }
}

// ---------------------------------------------------------------------------
extern "C" void kernel_launch(void* const* d_in, const int* in_sizes, int n_in,
                              void* d_out, int out_size, void* d_ws, size_t ws_size,
                              hipStream_t stream) {
  const float* x_q  = (const float*)d_in[0];
  const float* x_kv = (const float*)d_in[1];
  const unsigned char* mask = (const unsigned char*)d_in[2];  // jnp bool -> 1 byte
  const float* Wq = (const float*)d_in[3];
  const float* bq = (const float*)d_in[4];
  const float* Wk = (const float*)d_in[5];
  const float* bk = (const float*)d_in[6];
  const float* Wv = (const float*)d_in[7];
  const float* bv = (const float*)d_in[8];
  float* out = (float*)d_out;

  // Workspace: Q(8MB) | K(8MB) | V^T(8MB) bf16
  __bf16* Qbf = (__bf16*)d_ws;
  __bf16* Kbf = Qbf + (size_t)B_ * S_ * NQK;
  __bf16* Vt  = Kbf + (size_t)B_ * S_ * NQK;

  const int proj_blocks = (B_ * S_ / 128) * (NQK / 64);   // 32*16 = 512
  proj_gemm_bf16<false><<<proj_blocks, 256, 0, stream>>>(x_q,  Wq, bq, Qbf);
  proj_gemm_bf16<false><<<proj_blocks, 256, 0, stream>>>(x_kv, Wk, bk, Kbf);
  proj_gemm_bf16<true ><<<proj_blocks, 256, 0, stream>>>(x_kv, Wv, bv, Vt);

  const int waves = B_ * H_ * (S_ / 16);                  // 4096 q-tiles
  flash_attn<<<waves / 4, 128, 0, stream>>>(Qbf, Kbf, Vt, mask, out);
}
